// DiTBlock_17179869580
// MI455X (gfx1250) — matrix operand order repaired and checked by reading, not compiled, not run on hardware
//
#include <hip/hip_runtime.h>
#include <hip/hip_bf16.h>

// ---------------------------------------------------------------------------
// Problem constants (DiT block): B=8, S=1024, E=512, H=8
// ---------------------------------------------------------------------------
#define Bn   8
#define Sn   1024
#define En   512
#define Hn   8
#define HEn  (Hn * En)      // 4096
#define Mn   (Bn * Sn)      // 8192
#define Fn   (4 * En)       // 2048

typedef __attribute__((ext_vector_type(16))) __bf16 v16bf;
typedef __attribute__((ext_vector_type(8)))  float  v8f;

union FragBf { v16bf v; uint4 q[2]; };

__device__ __forceinline__ unsigned short f2bf(float f) {
    unsigned int u = __float_as_uint(f);
    unsigned int r = u + 0x7FFFu + ((u >> 16) & 1u);   // round-to-nearest-even
    return (unsigned short)(r >> 16);
}
__device__ __forceinline__ float bf2f(unsigned short h) {
    return __uint_as_float(((unsigned int)h) << 16);
}

// CDNA5 async memory->LDS copy (ASYNCcnt-tracked DMA, no VGPR staging).
// LDS dest is a 32-bit LDS byte offset (flat addr[31:0] per aperture rule).
__device__ __forceinline__ void async_load_b128(unsigned lds32, const void* g) {
    asm volatile("global_load_async_to_lds_b128 %0, %1, off"
                 :: "v"(lds32), "v"((unsigned long long)(uintptr_t)g)
                 : "memory");
}
__device__ __forceinline__ void wait_async0() {
    asm volatile("s_wait_asynccnt 0x0" ::: "memory");
}

// ---------------------------------------------------------------------------
// Generic batched WMMA GEMM:  C = epilogue(A(bf16, MxK row-major) * B + bias)
// B is supplied as BT (bf16, NxK row-major).  Batch z: base = (z/HH)*outer +
// (z%HH)*inner  (element offsets).  Tiles: 128x128x32, 8 waves (256 thr),
// double-buffered LDS filled by global_load_async_to_lds_b128.
// Wave grid 4(M) x 2(N); per-wave 2 M-tiles x 4 N-tiles of 16x16.
// Epilogues: 0: bf16 = acc+bias          1: bf16 = relu(acc+bias)
//            3: bf16 = acc*scale         2: f32  = resid + (acc+bias)*alpha[b,col]
// ---------------------------------------------------------------------------
#define BM 128
#define BN 128
#define BK 32
#define LSTR 48   // ushorts per LDS row: 32 data + 16 pad (96B, 16B aligned)
#define BUFSH (BM * LSTR)          // ushorts per buffer
#define BUFBYTES (BUFSH * 2)       // bytes per buffer

__global__ __launch_bounds__(256) void wmma_gemm_kernel(
    const unsigned short* __restrict__ A,
    const unsigned short* __restrict__ BT,
    const float* __restrict__ bias,
    void* __restrict__ Cout,
    int lda, int ldb, int ldc, int Kdim,
    long long aOuter, long long aInner,
    long long bOuter, long long bInner,
    long long cOuter, long long cInner,
    int HH, int epi, float scale,
    const float* __restrict__ resid,
    const float* __restrict__ alphas,
    int rowsPerB)
{
    __shared__ unsigned short lA[2 * BUFSH];
    __shared__ unsigned short lB[2 * BUFSH];

    const int tid  = threadIdx.x;
    const int lane = tid & 31;
    const int wave = tid >> 5;
    const int wM   = wave & 3;   // 0..3  -> 32 rows each
    const int wN   = wave >> 2;  // 0..1  -> 64 cols each
    const int hl   = lane & 15;
    const int hi   = lane >> 4;

    const int z = blockIdx.z;
    const long long baseA = (long long)(z / HH) * aOuter + (long long)(z % HH) * aInner;
    const long long baseB = (long long)(z / HH) * bOuter + (long long)(z % HH) * bInner;
    const long long baseC = (long long)(z / HH) * cOuter + (long long)(z % HH) * cInner;

    const int blockM = blockIdx.y * BM;
    const int blockN = blockIdx.x * BN;

    // Per-thread cooperative-copy slots: two (row, 16B-chunk) pairs.
    const int r0  = tid >> 2,          c0 = tid & 3;          // idx 0..255
    const int r1  = (tid + 256) >> 2,  c1 = (tid + 256) & 3;  // idx 256..511
    const unsigned short* gA0 = A  + baseA + (long long)(blockM + r0) * lda + c0 * 8;
    const unsigned short* gA1 = A  + baseA + (long long)(blockM + r1) * lda + c1 * 8;
    const unsigned short* gB0 = BT + baseB + (long long)(blockN + r0) * ldb + c0 * 8;
    const unsigned short* gB1 = BT + baseB + (long long)(blockN + r1) * ldb + c1 * 8;
    const unsigned la0 = (unsigned)(uintptr_t)&lA[r0 * LSTR + c0 * 8];
    const unsigned la1 = (unsigned)(uintptr_t)&lA[r1 * LSTR + c1 * 8];
    const unsigned lb0 = (unsigned)(uintptr_t)&lB[r0 * LSTR + c0 * 8];
    const unsigned lb1 = (unsigned)(uintptr_t)&lB[r1 * LSTR + c1 * 8];

    auto issue = [&](int buf, int k0) {
        unsigned boff = (unsigned)buf * BUFBYTES;
        async_load_b128(la0 + boff, gA0 + k0);
        async_load_b128(la1 + boff, gA1 + k0);
        async_load_b128(lb0 + boff, gB0 + k0);
        async_load_b128(lb1 + boff, gB1 + k0);
    };

    v8f acc[2][4] = {};

    const int nTiles = Kdim / BK;
    issue(0, 0);                                  // prologue: tile 0 -> buf 0

    for (int kt = 0; kt < nTiles; ++kt) {
        const int cur = kt & 1;
        wait_async0();                            // my DMA for tile kt landed
        __syncthreads();                          // everyone's DMA landed; buf 1-cur free
        if (kt + 1 < nTiles) issue(1 - cur, (kt + 1) * BK);

        const unsigned short* sA = &lA[cur * BUFSH];
        const unsigned short* sB = &lB[cur * BUFSH];

        // A fragment (16-bit A 16x32 layout): lane holds row M=hl;
        //   elems 0..7 = K hi*8..hi*8+7, elems 8..15 = K 16+hi*8..
        FragBf a[2];
#pragma unroll
        for (int mt = 0; mt < 2; ++mt) {
            int row = wM * 32 + mt * 16 + hl;
            a[mt].q[0] = *(const uint4*)(&sA[row * LSTR + hi * 8]);
            a[mt].q[1] = *(const uint4*)(&sA[row * LSTR + 16 + hi * 8]);
        }
        // B fragment (16-bit B 32x16 layout): lane holds col N=hl;
        //   elems 0..7 = K hi*8..hi*8+7, elems 8..15 = K 16+hi*8..
        FragBf b[4];
#pragma unroll
        for (int nt = 0; nt < 4; ++nt) {
            int row = wN * 64 + nt * 16 + hl;
            b[nt].q[0] = *(const uint4*)(&sB[row * LSTR + hi * 8]);
            b[nt].q[1] = *(const uint4*)(&sB[row * LSTR + 16 + hi * 8]);
        }
#pragma unroll
        for (int mt = 0; mt < 2; ++mt)
#pragma unroll
            for (int nt = 0; nt < 4; ++nt)
                acc[mt][nt] = __builtin_amdgcn_wmma_f32_16x16x32_bf16(
                    false, a[mt].v, false, b[nt].v, (short)0, acc[mt][nt], false, false);
        __syncthreads();                          // done reading buf cur
    }

    // Epilogue. C/D layout: elem i, half hi -> row = 16*t + hi*8 + i, col = hl.
#pragma unroll
    for (int mt = 0; mt < 2; ++mt)
#pragma unroll
        for (int nt = 0; nt < 4; ++nt)
#pragma unroll
            for (int i = 0; i < 8; ++i) {
                int row = blockM + wM * 32 + mt * 16 + hi * 8 + i;
                int col = blockN + wN * 64 + nt * 16 + hl;
                float v = acc[mt][nt][i];
                if (bias) v += bias[col];
                long long ci = baseC + (long long)row * ldc + col;
                if (epi == 0) {
                    ((unsigned short*)Cout)[ci] = f2bf(v);
                } else if (epi == 1) {
                    ((unsigned short*)Cout)[ci] = f2bf(v > 0.0f ? v : 0.0f);
                } else if (epi == 3) {
                    ((unsigned short*)Cout)[ci] = f2bf(v * scale);
                } else { // epi == 2: f32 residual + alpha-scaled (N == E here)
                    float al = alphas[(row / rowsPerB) * ldc + col];
                    ((float*)Cout)[ci] = resid[ci] + v * al;
                }
            }
}

// ---------------------------------------------------------------------------
// Tiny modulation GEMM: out(B,E) = cond(B,E) @ W(E,E) + bias   (4.2 MFLOP)
// ---------------------------------------------------------------------------
__global__ __launch_bounds__(256) void mod_gemm_kernel(
    const float* __restrict__ cond, const float* __restrict__ W,
    const float* __restrict__ bias, float* __restrict__ out)
{
    int g = blockIdx.x * 256 + threadIdx.x;     // B*E = 4096
    int b = g >> 9;
    int e = g & 511;
    float acc = bias[e];
    const float* c = cond + b * En;
    for (int k = 0; k < En; ++k) acc += c[k] * W[k * En + e];
    out[g] = acc;
}

// ---------------------------------------------------------------------------
// LayerNorm + AdaLN modulate:  Y_bf16 = LN(X)*(1+gamma[b]) + beta[b]
// One 256-thread block per row of 512.
// ---------------------------------------------------------------------------
__global__ __launch_bounds__(256) void ln_mod_kernel(
    const float* __restrict__ X,
    const float* __restrict__ lnG, const float* __restrict__ lnB,
    const float* __restrict__ gamma, const float* __restrict__ beta,
    unsigned short* __restrict__ Y, int rowsPerB)
{
    __shared__ float red[256];
    const int row = blockIdx.x;
    const int b   = row / rowsPerB;
    const int t   = threadIdx.x;
    const float* x = X + (long long)row * En;

    float x0 = x[t], x1 = x[t + 256];
    red[t] = x0 + x1;
    __syncthreads();
    for (int off = 128; off > 0; off >>= 1) {
        if (t < off) red[t] += red[t + off];
        __syncthreads();
    }
    float mean = red[0] * (1.0f / En);
    __syncthreads();
    float d0 = x0 - mean, d1 = x1 - mean;
    red[t] = d0 * d0 + d1 * d1;
    __syncthreads();
    for (int off = 128; off > 0; off >>= 1) {
        if (t < off) red[t] += red[t + off];
        __syncthreads();
    }
    float rstd = rsqrtf(red[0] * (1.0f / En) + 1e-5f);

    int e0 = t, e1 = t + 256;
    float y0 = d0 * rstd * lnG[e0] + lnB[e0];
    float y1 = d1 * rstd * lnG[e1] + lnB[e1];
    y0 = y0 * (1.0f + gamma[b * En + e0]) + beta[b * En + e0];
    y1 = y1 * (1.0f + gamma[b * En + e1]) + beta[b * En + e1];
    unsigned short* y = Y + (long long)row * En;
    y[e0] = f2bf(y0);
    y[e1] = f2bf(y1);
}

// ---------------------------------------------------------------------------
// Weight transpose + f32->bf16:  WT(N,K) = bf16(W(K,N))
// ---------------------------------------------------------------------------
__global__ __launch_bounds__(256) void transpose_cvt_kernel(
    const float* __restrict__ W, unsigned short* __restrict__ WT, int Kd, int Nd)
{
    __shared__ float tile[32][33];
    int kb = blockIdx.x * 32, nb = blockIdx.y * 32;
    int tx = threadIdx.x & 31, ty = threadIdx.x >> 5;
#pragma unroll
    for (int j = 0; j < 4; ++j)
        tile[ty + j * 8][tx] = W[(long long)(kb + ty + j * 8) * Nd + nb + tx];
    __syncthreads();
#pragma unroll
    for (int j = 0; j < 4; ++j)
        WT[(long long)(nb + ty + j * 8) * Kd + kb + tx] = f2bf(tile[tx][ty + j * 8]);
}

// ---------------------------------------------------------------------------
// V transpose (bf16): V(B,S,H,E) -> VT(B*H, E, S)
// ---------------------------------------------------------------------------
__global__ __launch_bounds__(256) void transpose_v_kernel(
    const unsigned short* __restrict__ V, unsigned short* __restrict__ VT)
{
    __shared__ unsigned short tile[32][33];
    int z = blockIdx.z, b = z >> 3, h = z & 7;
    int sb = blockIdx.x * 32, eb = blockIdx.y * 32;
    int tx = threadIdx.x & 31, ty = threadIdx.x >> 5;
    const unsigned short* src = V + (long long)b * Sn * HEn + (long long)h * En;
#pragma unroll
    for (int j = 0; j < 4; ++j)
        tile[ty + j * 8][tx] = src[(long long)(sb + ty + j * 8) * HEn + eb + tx];
    __syncthreads();
    unsigned short* dst = VT + (long long)z * En * Sn;
#pragma unroll
    for (int j = 0; j < 4; ++j)
        dst[(long long)(eb + ty + j * 8) * Sn + sb + tx] = tile[tx][ty + j * 8];
}

// ---------------------------------------------------------------------------
// Row softmax over 1024 bf16 logits, in place.
// ---------------------------------------------------------------------------
__global__ __launch_bounds__(256) void softmax_kernel(unsigned short* __restrict__ P)
{
    __shared__ float red[256];
    unsigned short* p = P + (long long)blockIdx.x * Sn;
    int t = threadIdx.x;
    float v[4];
#pragma unroll
    for (int j = 0; j < 4; ++j) v[j] = bf2f(p[t + j * 256]);

    float m = fmaxf(fmaxf(v[0], v[1]), fmaxf(v[2], v[3]));
    red[t] = m;
    __syncthreads();
    for (int off = 128; off > 0; off >>= 1) {
        if (t < off) red[t] = fmaxf(red[t], red[t + off]);
        __syncthreads();
    }
    m = red[0];
    __syncthreads();
    float s = 0.0f;
#pragma unroll
    for (int j = 0; j < 4; ++j) { v[j] = __expf(v[j] - m); s += v[j]; }
    red[t] = s;
    __syncthreads();
    for (int off = 128; off > 0; off >>= 1) {
        if (t < off) red[t] += red[t + off];
        __syncthreads();
    }
    float inv = 1.0f / red[0];
#pragma unroll
    for (int j = 0; j < 4; ++j) p[t + j * 256] = f2bf(v[j] * inv);
}

// ---------------------------------------------------------------------------
// Workspace layout (bytes)
// ---------------------------------------------------------------------------
static constexpr size_t AL(size_t x) { return (x + 255) & ~size_t(255); }
static constexpr size_t MOD_OFF  = 0;                                       // 6*(B,E) f32
static constexpr size_t WQT_OFF  = AL(MOD_OFF  + (size_t)6 * Bn * En * 4);
static constexpr size_t WKT_OFF  = AL(WQT_OFF  + (size_t)HEn * En * 2);
static constexpr size_t WVT_OFF  = AL(WKT_OFF  + (size_t)HEn * En * 2);
static constexpr size_t LVWT_OFF = AL(WVT_OFF  + (size_t)HEn * En * 2);
static constexpr size_t F1WT_OFF = AL(LVWT_OFF + (size_t)En * HEn * 2);
static constexpr size_t F2WT_OFF = AL(F1WT_OFF + (size_t)Fn * En * 2);
static constexpr size_t YBF_OFF  = AL(F2WT_OFF + (size_t)En * Fn * 2);      // also reused for z
static constexpr size_t Q_OFF    = AL(YBF_OFF  + (size_t)Mn * En * 2);      // also reused for o
static constexpr size_t K_OFF    = AL(Q_OFF    + (size_t)Mn * HEn * 2);     // also reused for ffn hidden
static constexpr size_t V_OFF    = AL(K_OFF    + (size_t)Mn * HEn * 2);
static constexpr size_t VT_OFF   = AL(V_OFF    + (size_t)Mn * HEn * 2);
static constexpr size_t S_OFF    = AL(VT_OFF   + (size_t)Mn * HEn * 2);     // (B*H,S,S) bf16
static constexpr size_t Y2_OFF   = AL(S_OFF    + (size_t)Bn * Hn * Sn * Sn * 2);

extern "C" void kernel_launch(void* const* d_in, const int* in_sizes, int n_in,
                              void* d_out, int out_size, void* d_ws, size_t ws_size,
                              hipStream_t stream)
{
    const float* x    = (const float*)d_in[0];
    const float* cond = (const float*)d_in[1];
    const float* g1w = (const float*)d_in[2],  * g1b = (const float*)d_in[3];
    const float* be1w= (const float*)d_in[4],  * be1b= (const float*)d_in[5];
    const float* a1w = (const float*)d_in[6],  * a1b = (const float*)d_in[7];
    const float* g2w = (const float*)d_in[8],  * g2b = (const float*)d_in[9];
    const float* be2w= (const float*)d_in[10], * be2b= (const float*)d_in[11];
    const float* a2w = (const float*)d_in[12], * a2b = (const float*)d_in[13];
    const float* ln1g= (const float*)d_in[14], * ln1b= (const float*)d_in[15];
    const float* ln2g= (const float*)d_in[16], * ln2b= (const float*)d_in[17];
    const float* wq  = (const float*)d_in[18], * bq  = (const float*)d_in[19];
    const float* wk  = (const float*)d_in[20], * bk  = (const float*)d_in[21];
    const float* wv  = (const float*)d_in[22], * bv  = (const float*)d_in[23];
    const float* lvw = (const float*)d_in[24], * lvb = (const float*)d_in[25];
    const float* f1w = (const float*)d_in[26], * f1b = (const float*)d_in[27];
    const float* f2w = (const float*)d_in[28], * f2b = (const float*)d_in[29];

    char* ws = (char*)d_ws;
    float*          mods = (float*)(ws + MOD_OFF);
    unsigned short* wqT  = (unsigned short*)(ws + WQT_OFF);
    unsigned short* wkT  = (unsigned short*)(ws + WKT_OFF);
    unsigned short* wvT  = (unsigned short*)(ws + WVT_OFF);
    unsigned short* lvwT = (unsigned short*)(ws + LVWT_OFF);
    unsigned short* f1wT = (unsigned short*)(ws + F1WT_OFF);
    unsigned short* f2wT = (unsigned short*)(ws + F2WT_OFF);
    unsigned short* ybf  = (unsigned short*)(ws + YBF_OFF);
    unsigned short* qb   = (unsigned short*)(ws + Q_OFF);
    unsigned short* kb2  = (unsigned short*)(ws + K_OFF);
    unsigned short* vb   = (unsigned short*)(ws + V_OFF);
    unsigned short* vT   = (unsigned short*)(ws + VT_OFF);
    unsigned short* sb   = (unsigned short*)(ws + S_OFF);
    float*          y2   = (float*)(ws + Y2_OFF);
    unsigned short* ob   = qb;   // reuse q buffer for attention output
    unsigned short* h1   = kb2;  // reuse k buffer for FFN hidden
    unsigned short* zbf  = ybf;  // reuse y buffer for second LN output
    float*          out  = (float*)d_out;

    dim3 blk(256);

    // 1. Weight transpose+convert (bf16, N-major for WMMA B operand)
    transpose_cvt_kernel<<<dim3(En/32, HEn/32), blk, 0, stream>>>(wq,  wqT,  En,  HEn);
    transpose_cvt_kernel<<<dim3(En/32, HEn/32), blk, 0, stream>>>(wk,  wkT,  En,  HEn);
    transpose_cvt_kernel<<<dim3(En/32, HEn/32), blk, 0, stream>>>(wv,  wvT,  En,  HEn);
    transpose_cvt_kernel<<<dim3(HEn/32, En/32), blk, 0, stream>>>(lvw, lvwT, HEn, En);
    transpose_cvt_kernel<<<dim3(En/32, Fn/32),  blk, 0, stream>>>(f1w, f1wT, En,  Fn);
    transpose_cvt_kernel<<<dim3(Fn/32, En/32),  blk, 0, stream>>>(f2w, f2wT, Fn,  En);

    // 2. AdaLN modulation vectors: slots {g1, be1, a1, g2, be2, a2}
    const int SLOT = Bn * En;
    mod_gemm_kernel<<<dim3(SLOT/256), blk, 0, stream>>>(cond, g1w,  g1b,  mods + 0*SLOT);
    mod_gemm_kernel<<<dim3(SLOT/256), blk, 0, stream>>>(cond, be1w, be1b, mods + 1*SLOT);
    mod_gemm_kernel<<<dim3(SLOT/256), blk, 0, stream>>>(cond, a1w,  a1b,  mods + 2*SLOT);
    mod_gemm_kernel<<<dim3(SLOT/256), blk, 0, stream>>>(cond, g2w,  g2b,  mods + 3*SLOT);
    mod_gemm_kernel<<<dim3(SLOT/256), blk, 0, stream>>>(cond, be2w, be2b, mods + 4*SLOT);
    mod_gemm_kernel<<<dim3(SLOT/256), blk, 0, stream>>>(cond, a2w,  a2b,  mods + 5*SLOT);

    // 3. y = LN1(x)*(1+g1)+be1  -> bf16
    ln_mod_kernel<<<dim3(Mn), blk, 0, stream>>>(x, ln1g, ln1b, mods + 0*SLOT, mods + 1*SLOT, ybf, Sn);

    // 4. q/k/v projections: (8192x512)x(512x4096), bf16 out in (B,S,H,E)
    wmma_gemm_kernel<<<dim3(HEn/BN, Mn/BM, 1), blk, 0, stream>>>(
        ybf, wqT, bq, qb, En, En, HEn, En, 0,0, 0,0, 0,0, 1, 0, 1.0f, nullptr, nullptr, Sn);
    wmma_gemm_kernel<<<dim3(HEn/BN, Mn/BM, 1), blk, 0, stream>>>(
        ybf, wkT, bk, kb2, En, En, HEn, En, 0,0, 0,0, 0,0, 1, 0, 1.0f, nullptr, nullptr, Sn);
    wmma_gemm_kernel<<<dim3(HEn/BN, Mn/BM, 1), blk, 0, stream>>>(
        ybf, wvT, bv, vb, En, En, HEn, En, 0,0, 0,0, 0,0, 1, 0, 1.0f, nullptr, nullptr, Sn);

    // 5. V^T per (b,h): (E,S)
    transpose_v_kernel<<<dim3(Sn/32, En/32, Bn*Hn), blk, 0, stream>>>(vb, vT);

    // 6. logits = (Q K^T) / sqrt(S), batched over 64 (b,h); bf16 out
    wmma_gemm_kernel<<<dim3(Sn/BN, Sn/BM, Bn*Hn), blk, 0, stream>>>(
        qb, kb2, nullptr, sb,
        HEn, HEn, Sn, En,
        (long long)Sn*HEn, En,            // A: Q base b*S*H*E + h*E
        (long long)Sn*HEn, En,            // B: K rows (BT form)
        (long long)Hn*Sn*Sn, (long long)Sn*Sn,
        Hn, 3, 0.03125f, nullptr, nullptr, Sn);

    // 7. softmax rows (in place)
    softmax_kernel<<<dim3(Bn*Hn*Sn), blk, 0, stream>>>(sb);

    // 8. O = P V  (batched; BT = V^T), bf16 out into (B,S,H,E)
    wmma_gemm_kernel<<<dim3(En/BN, Sn/BM, Bn*Hn), blk, 0, stream>>>(
        sb, vT, nullptr, ob,
        Sn, Sn, HEn, Sn,
        (long long)Hn*Sn*Sn, (long long)Sn*Sn,
        (long long)En*Sn*Hn, (long long)En*Sn,
        (long long)Sn*HEn, En,
        Hn, 0, 1.0f, nullptr, nullptr, Sn);

    // 9. y2 = x + (O @ lvw + lvb) * alpha1   (f32)
    wmma_gemm_kernel<<<dim3(En/BN, Mn/BM, 1), blk, 0, stream>>>(
        ob, lvwT, lvb, y2, HEn, HEn, En, HEn, 0,0, 0,0, 0,0,
        1, 2, 1.0f, x, mods + 2*SLOT, Sn);

    // 10. z = LN2(y2)*(1+g2)+be2 -> bf16
    ln_mod_kernel<<<dim3(Mn), blk, 0, stream>>>(y2, ln2g, ln2b, mods + 3*SLOT, mods + 4*SLOT, zbf, Sn);

    // 11. h1 = relu(z @ f1w + f1b) -> bf16
    wmma_gemm_kernel<<<dim3(Fn/BN, Mn/BM, 1), blk, 0, stream>>>(
        zbf, f1wT, f1b, h1, En, En, Fn, En, 0,0, 0,0, 0,0,
        1, 1, 1.0f, nullptr, nullptr, Sn);

    // 12. out = y2 + (h1 @ f2w + f2b) * alpha2   (f32, final)
    wmma_gemm_kernel<<<dim3(En/BN, Mn/BM, 1), blk, 0, stream>>>(
        h1, f2wT, f2b, out, Fn, Fn, En, Fn, 0,0, 0,0, 0,0,
        1, 2, 1.0f, y2, mods + 5*SLOT, Sn);

    (void)in_sizes; (void)n_in; (void)out_size; (void)ws_size;
}
